// ENCAML_33217277067338
// MI455X (gfx1250) — compile-verified, matrix operand order checked by reading, not run on hardware
//
#include <hip/hip_runtime.h>
#include <hip/hip_bf16.h>

// ---------------------------------------------------------------------------
// Types for CDNA5 WMMA (wave32): v_wmma_f32_16x16x32_bf16
// ---------------------------------------------------------------------------
typedef __attribute__((ext_vector_type(16))) __bf16 v16bf;
typedef __attribute__((ext_vector_type(8)))  __bf16 v8bf;
typedef __attribute__((ext_vector_type(8)))  float  v8f;

union VF8 { v8f v; float f[8]; };

#define E_DIM   100
#define F_DIM   256
#define C_DIM   50
#define B_DIM   32
#define L_DIM   2500
// Padded sequence rows: 4 halo + overhang so the last L-tile's (l<=2559)
// k=9 patch with Kpad=928 stays in-bounds: 2559+9+8 -> 2576.
#define L_EXT   2576

// LDS tile stride (bf16 elems). 40*2 = 80 B/row -> 16B-aligned 8-elem chunks.
#define TS 40

// CDNA5 async global->LDS staging (inline asm; verified to assemble).
#define USE_ASYNC_LDS 1

// ---------------------------------------------------------------------------
// Native gfx1250 tanh (TRANS op) with fallback.
// ---------------------------------------------------------------------------
__device__ __forceinline__ float fast_tanh(float x) {
#if __has_builtin(__builtin_amdgcn_tanhf)
  return __builtin_amdgcn_tanhf(x);
#elif __has_builtin(__builtin_amdgcn_tanh_f32)
  return __builtin_amdgcn_tanh_f32(x);
#else
  return tanhf(x);
#endif
}

// ---------------------------------------------------------------------------
// 16x32 bf16 WMMA fragment from an LDS tile (CDNA5 ISA 7.12.2):
//   lanes 0-15  : elems 0..7 = K[0..7],   elems 8..15 = K[16..23]
//   lanes 16-31 : elems 0..7 = K[8..15],  elems 8..15 = K[24..31]
// ---------------------------------------------------------------------------
__device__ __forceinline__ v16bf make_frag(const __bf16* base, int stride,
                                           int row0, int coff, int lane) {
  int half = (lane >> 4) & 1;
  int r    = lane & 15;
  const __bf16* p = base + (size_t)(row0 + r) * stride + coff + half * 8;
  union { v16bf v; v8bf h[2]; } u;
  u.h[0] = *(const v8bf*)(p);
  u.h[1] = *(const v8bf*)(p + 16);
  return u.v;
}

__device__ __forceinline__ v8f wmma_bf16(v16bf a, v16bf b, v8f c) {
  return __builtin_amdgcn_wmma_f32_16x16x32_bf16(
      false, a, false, b, (short)0, c, false, false);
}

#if USE_ASYNC_LDS
// Async copy helpers: INST_OFFSET is added to BOTH lds and global addresses.
__device__ __forceinline__ void async_b128x2(const __bf16* gsrc, __bf16* lds) {
  unsigned l = (unsigned)(size_t)lds;
  unsigned long long g = (unsigned long long)(size_t)gsrc;
  asm volatile("global_load_async_to_lds_b128 %0, %1, off"
               :: "v"(l), "v"(g) : "memory");
  asm volatile("global_load_async_to_lds_b128 %0, %1, off offset:16"
               :: "v"(l), "v"(g) : "memory");
}
__device__ __forceinline__ void async_b64x2(const __bf16* gsrc, __bf16* lds) {
  unsigned l = (unsigned)(size_t)lds;
  unsigned long long g = (unsigned long long)(size_t)gsrc;
  asm volatile("global_load_async_to_lds_b64 %0, %1, off"
               :: "v"(l), "v"(g) : "memory");
  asm volatile("global_load_async_to_lds_b64 %0, %1, off offset:8"
               :: "v"(l), "v"(g) : "memory");
}
#define ASYNC_WAIT() asm volatile("s_wait_asynccnt 0x0" ::: "memory")
#else
#define ASYNC_WAIT()
#endif

// ---------------------------------------------------------------------------
// 1) Embedding gather into zero-padded bf16 sequence: xpad[B][L_EXT][E]
// ---------------------------------------------------------------------------
__global__ void embed_pad_kernel(const float* __restrict__ embed,
                                 const int* __restrict__ x,
                                 __bf16* __restrict__ xpad) {
  int r = blockIdx.x;           // 0..L_EXT-1
  int b = blockIdx.y;
  int e = threadIdx.x;          // 0..127 (only <E active)
  if (e >= E_DIM) return;
  float v = 0.0f;
  int pos = r - 4;
  if (pos >= 0 && pos < L_DIM) {
    int tok = x[b * L_DIM + pos];
    v = embed[(size_t)tok * E_DIM + e];
  }
  xpad[((size_t)b * L_EXT + r) * E_DIM + e] = (__bf16)v;
}

// ---------------------------------------------------------------------------
// 2) Conv weight repack: cwb[f][d*E + e] = cw[f][e][d], K zero-padded to Kpad.
// ---------------------------------------------------------------------------
__global__ void conv_w_prep(const float* __restrict__ cw,
                            __bf16* __restrict__ cwb, int ktaps, int Kpad) {
  int f = blockIdx.x;
  int K = ktaps * E_DIM;
  for (int kk = threadIdx.x; kk < Kpad; kk += blockDim.x) {
    float v = 0.0f;
    if (kk < K) {
      int d = kk / E_DIM, e = kk % E_DIM;
      v = cw[((size_t)f * E_DIM + e) * ktaps + d];
    }
    cwb[(size_t)f * Kpad + kk] = (__bf16)v;
  }
}

// ---------------------------------------------------------------------------
// 3) uw repack to bf16, rows zero-padded 50 -> 64: uwb[64][256]
// ---------------------------------------------------------------------------
__global__ void uw_prep(const float* __restrict__ uw, __bf16* __restrict__ uwb) {
  int i = blockIdx.x * 256 + threadIdx.x;
  int c = i >> 8, f = i & 255;
  uwb[i] = (__bf16)((c < C_DIM) ? uw[c * F_DIM + f] : 0.0f);
}

// ---------------------------------------------------------------------------
// 4) Implicit-GEMM conv + bias + tanh:  x1[b][f][l] (bf16)
//    Fully async-staged, double-buffered LDS, one barrier per K-step.
// ---------------------------------------------------------------------------
__global__ __launch_bounds__(256)
void conv_tanh_kernel(const __bf16* __restrict__ xpad,
                      const __bf16* __restrict__ cwb,
                      const float* __restrict__ cb,
                      __bf16* __restrict__ x1, int ktaps, int Kpad) {
  __shared__ __attribute__((aligned(16))) __bf16 As[2][128 * TS];
  __shared__ __attribute__((aligned(16))) __bf16 Bs[2][64 * TS];

  const int b    = blockIdx.z;
  const int f0   = blockIdx.y * 128;
  const int l0   = blockIdx.x * 64;
  const int tid  = threadIdx.x;
  const int wave = tid >> 5, lane = tid & 31;
  const int roff = 4 - (ktaps >> 1);

  const __bf16* xb = xpad + (size_t)b * L_EXT * E_DIM;

  // staging coordinates (loop-invariant)
  const int arow = tid >> 1, ahc = tid & 1;          // A: 32 B / thread
  const int bn = tid & 63, bchunk = tid >> 6;        // B: 16 B / thread
  const __bf16* asrc0 = cwb + (size_t)(f0 + arow) * Kpad + ahc * 16;
  const __bf16* bsrc0 = xb + (size_t)(l0 + bn + roff) * E_DIM + bchunk * 8;

  auto stage = [&](int kb, int sel) {
#if USE_ASYNC_LDS
    async_b128x2(asrc0 + kb, &As[sel][arow * TS + ahc * 16]);
    async_b64x2 (bsrc0 + kb, &Bs[sel][bn * TS + bchunk * 8]);
#else
    *(v8bf*)&As[sel][arow * TS + ahc * 16]     = *(const v8bf*)(asrc0 + kb);
    *(v8bf*)&As[sel][arow * TS + ahc * 16 + 8] = *(const v8bf*)(asrc0 + kb + 8);
    union { uint2 u2[2]; v8bf v; } tb;
    tb.u2[0] = *(const uint2*)(bsrc0 + kb);
    tb.u2[1] = *(const uint2*)(bsrc0 + kb + 4);
    *(v8bf*)&Bs[sel][bn * TS + bchunk * 8] = tb.v;
#endif
  };

  VF8 acc[4];
#pragma unroll
  for (int t = 0; t < 4; ++t) acc[t].v = (v8f){0,0,0,0,0,0,0,0};

  stage(0, 0);
  int sel = 0;
  for (int kb = 0; kb < Kpad; kb += 32) {
    ASYNC_WAIT();            // our async writes for buffer `sel` landed
    __syncthreads();         // everyone's writes for `sel` visible; reads of
                             // `sel^1` from previous step are complete
    if (kb + 32 < Kpad) stage(kb + 32, sel ^ 1);

    v16bf afr = make_frag(&As[sel][0], TS, wave * 16, 0, lane);
    v16bf bfr[4];
#pragma unroll
    for (int t = 0; t < 4; ++t) bfr[t] = make_frag(&Bs[sel][0], TS, t * 16, 0, lane);
#pragma unroll
    for (int t = 0; t < 4; ++t) acc[t].v = wmma_bf16(afr, bfr[t], acc[t].v);
    sel ^= 1;
  }

  // Epilogue: bias + native tanh, store bf16 x1[b][f][l]
  const int half = lane >> 4, col = lane & 15;
  float cbr[8];
#pragma unroll
  for (int r = 0; r < 8; ++r) cbr[r] = cb[f0 + wave * 16 + r + 8 * half];
#pragma unroll
  for (int t = 0; t < 4; ++t) {
    int l = l0 + t * 16 + col;
    if (l >= L_DIM) continue;
#pragma unroll
    for (int r = 0; r < 8; ++r) {
      int f = f0 + wave * 16 + r + 8 * half;
      float v = fast_tanh(acc[t].f[r] + cbr[r]);
      x1[((size_t)b * F_DIM + f) * L_DIM + l] = (__bf16)v;
    }
  }
}

// ---------------------------------------------------------------------------
// 5) Attention scores: s[b][c][l] = uw[c,:] . x1[b,:,l]   (M=64pad, K=256)
// ---------------------------------------------------------------------------
__global__ __launch_bounds__(256)
void score_kernel(const __bf16* __restrict__ uwb,
                  const __bf16* __restrict__ x1,
                  float* __restrict__ s) {
  __shared__ __attribute__((aligned(16))) __bf16 Au[64 * 264];
  __shared__ __attribute__((aligned(16))) __bf16 Bs[64 * TS];

  const int b    = blockIdx.z;
  const int l0   = blockIdx.x * 64;
  const int tid  = threadIdx.x;
  const int wave = tid >> 5, lane = tid & 31;
  const int msub = wave & 3, npair = (wave >> 2) * 2;

  // Load the whole A (uwb, 64x256) once.
  {
    int row = tid & 63, cb0 = (tid >> 6) * 64;
#pragma unroll
    for (int j = 0; j < 64; j += 8)
      *(v8bf*)&Au[row * 264 + cb0 + j] = *(const v8bf*)&uwb[row * F_DIM + cb0 + j];
  }

  const int bn = tid & 63, bchunk = tid >> 6;
  const __bf16* px0 =
      x1 + ((size_t)b * F_DIM + bchunk * 8) * L_DIM + l0 + bn;

  VF8 acc[2];
  acc[0].v = (v8f){0,0,0,0,0,0,0,0};
  acc[1].v = (v8f){0,0,0,0,0,0,0,0};

  for (int kb = 0; kb < F_DIM; kb += 32) {
    // Stage B: Bs[n][kk] = x1[b][kb+kk][l0+n]; branchless (x1 has slack).
    {
      const __bf16* px = px0 + (size_t)kb * L_DIM;
#pragma unroll
      for (int j = 0; j < 8; ++j)
        Bs[bn * TS + bchunk * 8 + j] = px[(size_t)j * L_DIM];
      if (kb + 32 < F_DIM)
        __builtin_prefetch(px + (size_t)32 * L_DIM, 0, 0);
    }
    __syncthreads();

    v16bf afr = make_frag(Au, 264, msub * 16, kb, lane);
    v16bf bfr[2];
#pragma unroll
    for (int u = 0; u < 2; ++u) bfr[u] = make_frag(Bs, TS, (npair + u) * 16, 0, lane);
#pragma unroll
    for (int u = 0; u < 2; ++u) acc[u].v = wmma_bf16(afr, bfr[u], acc[u].v);
    __syncthreads();
  }

  const int half = lane >> 4, col = lane & 15;
#pragma unroll
  for (int u = 0; u < 2; ++u) {
    int l = l0 + (npair + u) * 16 + col;
    if (l >= L_DIM) continue;
#pragma unroll
    for (int r = 0; r < 8; ++r) {
      int c = msub * 16 + r + 8 * half;
      if (c < C_DIM)
        s[((size_t)b * C_DIM + c) * L_DIM + l] = acc[u].f[r];
    }
  }
}

// ---------------------------------------------------------------------------
// 6) Softmax over L per (b,c); alpha written straight into d_out.
// ---------------------------------------------------------------------------
__global__ __launch_bounds__(256)
void softmax_kernel(const float* __restrict__ s, float* __restrict__ alpha) {
  const int bc = blockIdx.x;
  const float* row = s + (size_t)bc * L_DIM;
  float* out = alpha + (size_t)bc * L_DIM;
  __shared__ float red[256];
  const int tid = threadIdx.x;

  float lmax = -3.0e38f;
  for (int l = tid; l < L_DIM; l += 256) lmax = fmaxf(lmax, row[l]);
  red[tid] = lmax; __syncthreads();
  for (int st = 128; st > 0; st >>= 1) {
    if (tid < st) red[tid] = fmaxf(red[tid], red[tid + st]);
    __syncthreads();
  }
  const float m = red[0]; __syncthreads();

  float lsum = 0.0f;
  for (int l = tid; l < L_DIM; l += 256) lsum += __expf(row[l] - m);
  red[tid] = lsum; __syncthreads();
  for (int st = 128; st > 0; st >>= 1) {
    if (tid < st) red[tid] += red[tid + st];
    __syncthreads();
  }
  const float inv = 1.0f / red[0];

  for (int l = tid; l < L_DIM; l += 256)
    out[l] = __expf(row[l] - m) * inv;
}

// ---------------------------------------------------------------------------
// 7) Attention pooling: m[b][c][f] = sum_l alpha[b][c][l] * x1[b][f][l]
//    M=64(pad C), N=64 (f tile), K=2528; async-B, double-buffered LDS.
// ---------------------------------------------------------------------------
#define KLAST 2496   // last K-step start (79 steps of 32; tail guarded)

__global__ __launch_bounds__(256)
void mpool_kernel(const float* __restrict__ alpha,
                  const __bf16* __restrict__ x1,
                  float* __restrict__ mbuf) {
  __shared__ __attribute__((aligned(16))) __bf16 As[2][64 * TS];
  __shared__ __attribute__((aligned(16))) __bf16 Bs[2][64 * TS];

  const int b    = blockIdx.z;
  const int f0   = blockIdx.x * 64;
  const int tid  = threadIdx.x;
  const int wave = tid >> 5, lane = tid & 31;
  const int msub = wave & 3, npair = (wave >> 2) * 2;

  const int ci = tid & 63, chunk = tid >> 6;
  const float*  pa0 = alpha + ((size_t)b * C_DIM + ci) * L_DIM + chunk * 8;
  const __bf16* pb0 = x1 + ((size_t)b * F_DIM + f0 + ci) * L_DIM + chunk * 8;

  auto stage = [&](int kb, int sel) {
    // B (x1): async copy, branchless (x1 has tail slack)
#if USE_ASYNC_LDS
    async_b64x2(pb0 + kb, &Bs[sel][ci * TS + chunk * 8]);
#else
    union { uint2 u2[2]; v8bf v; } tb;
    tb.u2[0] = *(const uint2*)(pb0 + kb);
    tb.u2[1] = *(const uint2*)(pb0 + kb + 4);
    *(v8bf*)&Bs[sel][ci * TS + chunk * 8] = tb.v;
#endif
    // A (alpha f32 -> bf16): c guard is thread-uniform; tail step guarded
    if (ci < C_DIM) {
      const float* pa = pa0 + kb;
      if (kb + 31 < L_DIM) {
#pragma unroll
        for (int j = 0; j < 8; ++j)
          As[sel][ci * TS + chunk * 8 + j] = (__bf16)pa[j];
      } else {
#pragma unroll
        for (int j = 0; j < 8; ++j) {
          int l = kb + chunk * 8 + j;
          As[sel][ci * TS + chunk * 8 + j] =
              (l < L_DIM) ? (__bf16)pa[j] : (__bf16)0.0f;
        }
      }
    } else {
#pragma unroll
      for (int j = 0; j < 8; ++j)
        As[sel][ci * TS + chunk * 8 + j] = (__bf16)0.0f;
    }
  };

  VF8 acc[2];
  acc[0].v = (v8f){0,0,0,0,0,0,0,0};
  acc[1].v = (v8f){0,0,0,0,0,0,0,0};

  stage(0, 0);
  int sel = 0;
  for (int kb = 0; kb <= KLAST; kb += 32) {
    ASYNC_WAIT();
    __syncthreads();
    if (kb < KLAST) stage(kb + 32, sel ^ 1);

    v16bf afr = make_frag(&As[sel][0], TS, msub * 16, 0, lane);
    v16bf bfr[2];
#pragma unroll
    for (int u = 0; u < 2; ++u)
      bfr[u] = make_frag(&Bs[sel][0], TS, (npair + u) * 16, 0, lane);
#pragma unroll
    for (int u = 0; u < 2; ++u) acc[u].v = wmma_bf16(afr, bfr[u], acc[u].v);
    sel ^= 1;
  }

  const int half = lane >> 4, col = lane & 15;
#pragma unroll
  for (int u = 0; u < 2; ++u) {
    int f = f0 + (npair + u) * 16 + col;
#pragma unroll
    for (int r = 0; r < 8; ++r) {
      int c = msub * 16 + r + 8 * half;
      if (c < C_DIM)
        mbuf[((size_t)b * C_DIM + c) * F_DIM + f] = acc[u].f[r];
    }
  }
}

// ---------------------------------------------------------------------------
// 8) Final diagonal layer: y[b][c] = fb[c] + sum_{i,f} fw[c][i*F+f]*m_i[b][c][f]
// ---------------------------------------------------------------------------
__global__ void final_kernel(const float* __restrict__ fw,
                             const float* __restrict__ fb,
                             const float* __restrict__ mbuf,
                             float* __restrict__ y) {
  int idx = blockIdx.x * 256 + threadIdx.x;
  if (idx >= B_DIM * C_DIM) return;
  int b = idx / C_DIM, c = idx % C_DIM;
  float accv = fb[c];
  for (int i = 0; i < 4; ++i) {
    const float* mrow = mbuf + (((size_t)i * B_DIM + b) * C_DIM + c) * F_DIM;
    const float* frow = fw + (size_t)c * (4 * F_DIM) + i * F_DIM;
    for (int f = 0; f < F_DIM; ++f) accv += frow[f] * mrow[f];
  }
  y[idx] = accv;
}

// ---------------------------------------------------------------------------
// Host launcher
// ---------------------------------------------------------------------------
extern "C" void kernel_launch(void* const* d_in, const int* in_sizes, int n_in,
                              void* d_out, int out_size, void* d_ws, size_t ws_size,
                              hipStream_t stream) {
  (void)in_sizes; (void)n_in; (void)out_size; (void)ws_size;

  const float* embed = (const float*)d_in[0];
  const int*   x     = (const int*)d_in[1];
  const float* cwv[4] = {(const float*)d_in[2],  (const float*)d_in[5],
                         (const float*)d_in[8],  (const float*)d_in[11]};
  const float* cbv[4] = {(const float*)d_in[3],  (const float*)d_in[6],
                         (const float*)d_in[9],  (const float*)d_in[12]};
  const float* uwv[4] = {(const float*)d_in[4],  (const float*)d_in[7],
                         (const float*)d_in[10], (const float*)d_in[13]};
  const float* fw = (const float*)d_in[14];
  const float* fb = (const float*)d_in[15];

  float* y = (float*)d_out;
  float* alpha_base = y + B_DIM * C_DIM;

  // ---- workspace carve-up (256B aligned) ----
  char* ws = (char*)d_ws;
  size_t off = 0;
  auto carve = [&](size_t bytes) {
    char* p = ws + off;
    off += (bytes + 255) & ~(size_t)255;
    return (void*)p;
  };
  __bf16* xpad = (__bf16*)carve((size_t)B_DIM * L_EXT * E_DIM * 2);      // 16.49 MB
  __bf16* cwb  = (__bf16*)carve((size_t)F_DIM * 928 * 2);                // 0.48 MB
  __bf16* uwb  = (__bf16*)carve((size_t)64 * F_DIM * 2);                 // 32 KB
  __bf16* x1   = (__bf16*)carve((size_t)B_DIM * F_DIM * L_DIM * 2 + 512);// 40.96 MB (+tile slack)
  float*  sbuf = (float*) carve((size_t)B_DIM * C_DIM * L_DIM * 4);      // 16.0 MB
  float*  mbuf = (float*) carve((size_t)4 * B_DIM * C_DIM * F_DIM * 4);  // 6.55 MB

  const int KT[4] = {3, 5, 7, 9};
  const int LT = (L_DIM + 63) / 64;   // 40 L-tiles

  embed_pad_kernel<<<dim3(L_EXT, B_DIM), 128, 0, stream>>>(embed, x, xpad);

  for (int i = 0; i < 4; ++i) {
    const int kt = KT[i];
    const int Kpad = (kt * E_DIM + 31) & ~31;

    conv_w_prep<<<F_DIM, 128, 0, stream>>>(cwv[i], cwb, kt, Kpad);
    conv_tanh_kernel<<<dim3(LT, F_DIM / 128, B_DIM), 256, 0, stream>>>(
        xpad, cwb, cbv[i], x1, kt, Kpad);

    uw_prep<<<64, 256, 0, stream>>>(uwv[i], uwb);
    score_kernel<<<dim3(LT, 1, B_DIM), 256, 0, stream>>>(uwb, x1, sbuf);

    float* alpha_i = alpha_base + (size_t)i * B_DIM * C_DIM * L_DIM;
    softmax_kernel<<<B_DIM * C_DIM, 256, 0, stream>>>(sbuf, alpha_i);

    mpool_kernel<<<dim3(F_DIM / 64, 1, B_DIM), 256, 0, stream>>>(
        alpha_i, x1, mbuf + (size_t)i * B_DIM * C_DIM * F_DIM);
  }

  final_kernel<<<(B_DIM * C_DIM + 255) / 256, 256, 0, stream>>>(fw, fb, mbuf, y);
}
